// EvoformerStack_58918361366841
// MI455X (gfx1250) — compile-verified, matrix operand order checked by reading, not run on hardware
//
#include <hip/hip_runtime.h>
#include <math.h>

// ---------------------------------------------------------------------------
// Types for CDNA5 WMMA (wave32, 16x16x32 f16 -> f32 accumulate)
// ---------------------------------------------------------------------------
typedef __attribute__((ext_vector_type(16))) _Float16 v16h;
typedef __attribute__((ext_vector_type(8)))  _Float16 v8h;
typedef __attribute__((ext_vector_type(8)))  float    v8f;

#define BM 64
#define BK 32
#define LDSPAD 8   // pad rows to 40 halves = 80B so 8-half groups stay 16B aligned

// Model constants
#define CC   128
#define LL   256
#define RR   64
#define HMM  8
#define DMM  32
#define HPP  4
#define DPP  32
#define RM   (RR*LL)    // 16384 rows of m
#define ZR   (LL*LL)    // 65536 rows of z

// ---------------------------------------------------------------------------
// Generic strided batched GEMM with WMMA.
// D[b1,b2][m,n] (+)= act( scale * sum_{k2,k} A[m, k2,k] * B[k2,k, n] + bias[n] )
// TN = B-subtiles per wave (1 -> 64x32 block, 2 -> 64x64 block, 2 wmma/step)
// ---------------------------------------------------------------------------
struct GemmP {
  const float* __restrict__ A;
  const float* __restrict__ B;
  const float* __restrict__ bias;
  float* __restrict__ D;
  int M, N, K, K2;
  long sAm, sAk, sAk2;
  long sBk, sBn, sBk2;
  long sDm, sDn;
  int  nb2;                       // grid.z = nb1*nb2 ; b1 = z/nb2, b2 = z%nb2
  long bA1, bA2, bB1, bB2, bD1, bD2;
  float scale;
  int act;                        // 0 none, 1 relu, 2 sigmoid
  int accum;                      // 1: D += result
  int biasStride;
};

union PackU { _Float16 h[2]; unsigned u; };

template <int TN>
__global__ __launch_bounds__(256) void k_gemm_wmma(GemmP p) {
  constexpr int BNv = 32 * TN;
  // Ping-pong LDS tiles: one barrier per K-step
  __shared__ __align__(16) _Float16 Ah[2][BM][BK + LDSPAD];
  __shared__ __align__(16) _Float16 Bh[2][BNv][BK + LDSPAD];

  const int tid  = threadIdx.x;
  const int lane = tid & 31;
  const int wave = tid >> 5;
  const int wm   = wave & 3;      // 4 waves along M
  const int wn   = wave >> 2;     // 2 waves along N

  const int bz = blockIdx.z;
  const int b1 = bz / p.nb2;
  const int b2 = bz % p.nb2;
  const float* __restrict__ Ab = p.A + (long)b1 * p.bA1 + (long)b2 * p.bA2;
  const float* __restrict__ Bb = p.B + (long)b1 * p.bB1 + (long)b2 * p.bB2;
  float*       __restrict__ Db = p.D + (long)b1 * p.bD1 + (long)b2 * p.bD2;

  const int m0 = blockIdx.y * BM;
  const int n0 = blockIdx.x * BNv;

  v8f acc[TN];
#pragma unroll
  for (int t = 0; t < TN; ++t) acc[t] = (v8f){0.f,0.f,0.f,0.f,0.f,0.f,0.f,0.f};

  const int mrow = lane & 15;
  const int kb   = (lane >> 4) * 8;   // 0 or 8

  // ---- Loop-invariant staging coordinates (K-pair granularity) ----
  // A tile: 64 rows x 16 k-pairs = 1024 pairs -> 4 per thread
  int  rowA[4], kpA[4]; long offAm[4]; bool okm[4];
#pragma unroll
  for (int r = 0; r < 4; ++r) {
    int pr = tid + r * 256;
    rowA[r] = pr >> 4;
    kpA[r]  = (pr & 15) * 2;
    int gm  = m0 + rowA[r];
    okm[r]  = gm < p.M;
    offAm[r] = (long)(okm[r] ? gm : (p.M - 1)) * p.sAm;
  }
  // B tile: BNv rows x 16 k-pairs -> 2*TN per thread
  int  rowB[2*TN], kpB[2*TN]; long offBn[2*TN]; bool okn[2*TN];
#pragma unroll
  for (int r = 0; r < 2 * TN; ++r) {
    int pr = tid + r * 256;
    rowB[r] = pr >> 4;
    kpB[r]  = (pr & 15) * 2;
    int gn  = n0 + rowB[r];
    okn[r]  = gn < p.N;
    offBn[r] = (long)(okn[r] ? gn : (p.N - 1)) * p.sBn;
  }

  int par = 0;

  for (int k2 = 0; k2 < p.K2; ++k2) {
    const float* __restrict__ A2 = Ab + (long)k2 * p.sAk2;
    const float* __restrict__ B2 = Bb + (long)k2 * p.sBk2;

    auto step = [&](int k0, int pb) {
      // ---- Gather loads: indices clamped in-bounds, values zeroed later ----
      float fa0[4], fa1[4]; bool oka[4];
#pragma unroll
      for (int r = 0; r < 4; ++r) {
        int gk = k0 + kpA[r];
        bool okk = gk < p.K;            // K is even; gk even -> gk+1 also ok
        oka[r] = okm[r] && okk;
        long ka = (long)(okk ? gk : 0) * p.sAk;
        const float* pa = A2 + offAm[r] + ka;
        fa0[r] = pa[0];
        fa1[r] = pa[p.sAk];
      }
      float fb0[2*TN], fb1[2*TN]; bool okb[2*TN];
#pragma unroll
      for (int r = 0; r < 2 * TN; ++r) {
        int gk = k0 + kpB[r];
        bool okk = gk < p.K;
        okb[r] = okn[r] && okk;
        long kbo = (long)(okk ? gk : 0) * p.sBk;
        const float* pb2 = B2 + offBn[r] + kbo;
        fb0[r] = pb2[0];
        fb1[r] = pb2[p.sBk];
      }

      // ---- Packed f16 stores into ping-pong LDS tile ----
#pragma unroll
      for (int r = 0; r < 4; ++r) {
        PackU u;
        u.h[0] = oka[r] ? (_Float16)fa0[r] : (_Float16)0.0f;
        u.h[1] = oka[r] ? (_Float16)fa1[r] : (_Float16)0.0f;
        *reinterpret_cast<unsigned*>(&Ah[pb][rowA[r]][kpA[r]]) = u.u;
      }
#pragma unroll
      for (int r = 0; r < 2 * TN; ++r) {
        PackU u;
        u.h[0] = okb[r] ? (_Float16)fb0[r] : (_Float16)0.0f;
        u.h[1] = okb[r] ? (_Float16)fb1[r] : (_Float16)0.0f;
        *reinterpret_cast<unsigned*>(&Bh[pb][rowB[r]][kpB[r]]) = u.u;
      }
      __syncthreads();

      // ---- Per-lane fragments: two 16B LDS loads each, A reused TN times ----
      const v8h a_lo = *(const v8h*)&Ah[pb][wm * 16 + mrow][kb];
      const v8h a_hi = *(const v8h*)&Ah[pb][wm * 16 + mrow][16 + kb];
      v16h a;
#pragma unroll
      for (int i = 0; i < 8; ++i) { a[i] = a_lo[i]; a[8 + i] = a_hi[i]; }
#pragma unroll
      for (int t = 0; t < TN; ++t) {
        const int brow = wn * (16 * TN) + t * 16 + mrow;
        const v8h b_lo = *(const v8h*)&Bh[pb][brow][kb];
        const v8h b_hi = *(const v8h*)&Bh[pb][brow][16 + kb];
        v16h bfr;
#pragma unroll
        for (int i = 0; i < 8; ++i) { bfr[i] = b_lo[i]; bfr[8 + i] = b_hi[i]; }
        acc[t] = __builtin_amdgcn_wmma_f32_16x16x32_f16(false, a, false, bfr,
                                                        (short)0, acc[t], false, false);
      }
    };

    int k0 = 0;
    for (; k0 + 2 * BK <= p.K; k0 += 2 * BK) {
      step(k0, par); par ^= 1;
      step(k0 + BK, par); par ^= 1;
    }
    for (; k0 < p.K; k0 += BK) { step(k0, par); par ^= 1; }
  }

  // Epilogue: C/D layout — lane n = lane&15, rows m = v + 8*(lane>=16)
  const int mbase = m0 + wm * 16 + ((lane >> 4) ? 8 : 0);
#pragma unroll
  for (int t = 0; t < TN; ++t) {
    const int n = n0 + wn * (16 * TN) + t * 16 + (lane & 15);
    if (n < p.N) {
      float bv = p.bias ? p.bias[(long)n * p.biasStride] : 0.f;
#pragma unroll
      for (int v = 0; v < 8; ++v) {
        int m = mbase + v;
        if (m < p.M) {
          long off = (long)m * p.sDm + (long)n * p.sDn;
          float val = acc[t][v] * p.scale + bv;
          if (p.act == 1) val = val > 0.f ? val : 0.f;
          else if (p.act == 2) val = 1.f / (1.f + __expf(-val));
          if (p.accum) val += Db[off];
          Db[off] = val;
        }
      }
    }
  }
}

// ---------------------------------------------------------------------------
// LayerNorm over last dim (C = 128), one row per 128-thread block
// ---------------------------------------------------------------------------
__global__ __launch_bounds__(128) void k_ln(const float* __restrict__ x,
                                            const float* __restrict__ g,
                                            const float* __restrict__ b,
                                            float* __restrict__ y, int rows) {
  int row = blockIdx.x;
  if (row >= rows) return;
  int t = threadIdx.x;
  float v = x[(long)row * CC + t];
  __shared__ float red[128];
  red[t] = v; __syncthreads();
  for (int s = 64; s > 0; s >>= 1) { if (t < s) red[t] += red[t + s]; __syncthreads(); }
  float mu = red[0] * (1.f / CC);
  __syncthreads();
  float d = v - mu;
  red[t] = d * d; __syncthreads();
  for (int s = 64; s > 0; s >>= 1) { if (t < s) red[t] += red[t + s]; __syncthreads(); }
  float var = red[0] * (1.f / CC);
  float r = rsqrtf(var + 1e-5f);
  y[(long)row * CC + t] = d * r * g[t] + b[t];
}

// ---------------------------------------------------------------------------
// Softmax over last dim with fused scale + strided additive bias.
// rows are (i1,i2,i3); bias index = i1*bs1 + i2*bs2 + i3*bs3 + col*bsc
// ---------------------------------------------------------------------------
struct SoftP {
  float* s; const float* bias;
  int n1, n2, n3, cols;
  long bs1, bs2, bs3, bsc;
  float scale;
};

__global__ __launch_bounds__(128) void k_softmax(SoftP p) {
  long row = blockIdx.x;
  int i3 = (int)(row % p.n3); long r2 = row / p.n3;
  int i2 = (int)(r2 % p.n2);  int  i1 = (int)(r2 / p.n2);
  float* srow = p.s + row * (long)p.cols;
  const float* bb = p.bias
      ? p.bias + (long)i1 * p.bs1 + (long)i2 * p.bs2 + (long)i3 * p.bs3
      : nullptr;
  int t = threadIdx.x;
  __shared__ float red[128];

  float mx = -3.4e38f;
  for (int c = t; c < p.cols; c += 128) {
    float v = srow[c] * p.scale + (bb ? bb[(long)c * p.bsc] : 0.f);
    srow[c] = v;
    mx = fmaxf(mx, v);
  }
  red[t] = mx; __syncthreads();
  for (int s = 64; s > 0; s >>= 1) { if (t < s) red[t] = fmaxf(red[t], red[t + s]); __syncthreads(); }
  mx = red[0]; __syncthreads();

  float sum = 0.f;
  for (int c = t; c < p.cols; c += 128) { float e = __expf(srow[c] - mx); srow[c] = e; sum += e; }
  red[t] = sum; __syncthreads();
  for (int s = 64; s > 0; s >>= 1) { if (t < s) red[t] += red[t + s]; __syncthreads(); }
  float inv = 1.f / red[0];
  for (int c = t; c < p.cols; c += 128) srow[c] *= inv;
}

// ---------------------------------------------------------------------------
// Pointwise kernels
// ---------------------------------------------------------------------------
__global__ void k_mul(float* dst, const float* a, const float* b, long n) {
  for (long i = (long)blockIdx.x * 256 + threadIdx.x; i < n; i += (long)gridDim.x * 256)
    dst[i] = a[i] * b[i];
}
__global__ void k_muladd(float* dst, const float* a, const float* b, long n) {
  for (long i = (long)blockIdx.x * 256 + threadIdx.x; i < n; i += (long)gridDim.x * 256)
    dst[i] += a[i] * b[i];
}

struct CopyP {
  const float* src; float* dst;
  int n0, n1, n2;
  long ss0, ss1, ss2, ds0, ds1, ds2;
};
__global__ void k_copy3(CopyP p) {
  long total = (long)p.n0 * p.n1 * p.n2;
  for (long i = (long)blockIdx.x * 256 + threadIdx.x; i < total; i += (long)gridDim.x * 256) {
    int i2 = (int)(i % p.n2); long t = i / p.n2;
    int i1 = (int)(t % p.n1); int i0 = (int)(t / p.n1);
    p.dst[(long)i0 * p.ds0 + (long)i1 * p.ds1 + (long)i2 * p.ds2] =
        p.src[(long)i0 * p.ss0 + (long)i1 * p.ss1 + (long)i2 * p.ss2];
  }
}

// ---------------------------------------------------------------------------
// Host-side helpers
// ---------------------------------------------------------------------------
static inline GemmP gp0() {
  GemmP g{};
  g.K2 = 1; g.nb2 = 1; g.scale = 1.f; g.biasStride = 1;
  g.sDn = 1;
  return g;
}
static inline void launch_gemm(hipStream_t st, const GemmP& g, int nb1) {
  if ((g.N & 63) == 0) {
    dim3 grid(g.N / 64, (g.M + BM - 1) / BM, nb1 * g.nb2);
    k_gemm_wmma<2><<<grid, 256, 0, st>>>(g);
  } else {
    dim3 grid((g.N + 31) / 32, (g.M + BM - 1) / BM, nb1 * g.nb2);
    k_gemm_wmma<1><<<grid, 256, 0, st>>>(g);
  }
}
static inline void launch_soft(hipStream_t st, const SoftP& s) {
  long rows = (long)s.n1 * s.n2 * s.n3;
  k_softmax<<<(unsigned)rows, 128, 0, st>>>(s);
}
static inline unsigned pwgrid(long n) {
  long g = (n + 255) / 256; if (g > 32768) g = 32768; return (unsigned)g;
}

// Parameter views (sorted-key flattening order of the python dicts)
struct AttnP     { const float *gb,*gw,*nb,*ng,*ob,*ow,*pair,*qkv; };
struct AttnNPP   { const float *gb,*gw,*nb,*ng,*ob,*ow,*qkv; };
struct TransP    { const float *b1,*b2,*nb,*ng,*w1,*w2; };
struct OpmP      { const float *b,*nb,*ng,*ob,*ow,*w; };
struct TriMulP   { const float *b1,*b2,*gb,*gw,*n2b,*n2g,*nb,*ng,*ob,*ow,*w1,*w2; };
struct BlockP    { AttnNPP col; TransP mt; OpmP opm; TransP pt;
                   AttnP row; AttnP tca; TriMulP tmi; TriMulP tmo; AttnP tra; };

struct WS {
  float *zln, *mln, *bigA, *bigB, *genA, *genB, *genC, *smallA, *smallB;
};

// --------------------------- op implementations ----------------------------
static void op_msa_row(hipStream_t st, const AttnP& P, float* Mres, float* Zres, const WS& w) {
  k_ln<<<RM, 128, 0, st>>>(Mres, P.ng, P.nb, w.mln, RM);
  { GemmP g = gp0(); g.A=w.mln; g.B=P.qkv; g.D=w.bigA;            // qkv [RM,768]
    g.M=RM; g.N=3*HMM*DMM; g.K=CC; g.sAm=CC; g.sAk=1; g.sBk=3*HMM*DMM; g.sBn=1; g.sDm=3*HMM*DMM;
    launch_gemm(st, g, 1); }
  { GemmP g = gp0(); g.A=Zres; g.B=P.pair; g.D=w.smallA;          // pair bias [ZR,8]
    g.M=ZR; g.N=HMM; g.K=CC; g.sAm=CC; g.sAk=1; g.sBk=HMM; g.sBn=1; g.sDm=HMM;
    launch_gemm(st, g, 1); }
  { GemmP g = gp0();                                              // scores[h][i][j], sum over r
    g.A=w.bigA; g.B=w.bigA+HMM*DMM; g.D=w.genC;
    g.M=LL; g.N=LL; g.K=DMM; g.K2=RR;
    g.sAm=768; g.sAk=1; g.sAk2=(long)LL*768;
    g.sBk=1; g.sBn=768; g.sBk2=(long)LL*768;
    g.sDm=LL; g.bA1=DMM; g.bB1=DMM; g.bD1=(long)LL*LL;
    launch_gemm(st, g, HMM); }
  { SoftP s{}; s.s=w.genC; s.bias=w.smallA;
    s.n1=HMM; s.n2=LL; s.n3=1; s.cols=LL;
    s.bs1=1; s.bs2=(long)LL*HMM; s.bs3=0; s.bsc=HMM;
    s.scale=1.f/sqrtf((float)DMM); launch_soft(st, s); }
  { GemmP g = gp0();                                              // AV: out[(r*L+i), h*32+d]
    g.A=w.genC; g.B=w.bigA+2*HMM*DMM; g.D=w.genA;
    g.M=LL; g.N=DMM; g.K=LL;
    g.sAm=LL; g.sAk=1; g.sBk=768; g.sBn=1; g.sDm=HMM*DMM;
    g.nb2=RR;
    g.bA1=(long)LL*LL; g.bA2=0;
    g.bB1=DMM;         g.bB2=(long)LL*768;
    g.bD1=DMM;         g.bD2=(long)LL*HMM*DMM;
    launch_gemm(st, g, HMM); }
  { GemmP g = gp0(); g.A=w.mln; g.B=P.gw; g.bias=P.gb; g.D=w.genB; g.act=2;  // gate
    g.M=RM; g.N=HMM*DMM; g.K=CC; g.sAm=CC; g.sAk=1; g.sBk=HMM*DMM; g.sBn=1; g.sDm=HMM*DMM;
    launch_gemm(st, g, 1); }
  k_mul<<<pwgrid((long)RM*HMM*DMM), 256, 0, st>>>(w.genA, w.genA, w.genB, (long)RM*HMM*DMM);
  { GemmP g = gp0(); g.A=w.genA; g.B=P.ow; g.bias=P.ob; g.D=Mres; g.accum=1; // out proj
    g.M=RM; g.N=CC; g.K=HMM*DMM; g.sAm=HMM*DMM; g.sAk=1; g.sBk=CC; g.sBn=1; g.sDm=CC;
    launch_gemm(st, g, 1); }
}

static void op_msa_col(hipStream_t st, const AttnNPP& P, float* Mres, const WS& w) {
  k_ln<<<RM, 128, 0, st>>>(Mres, P.ng, P.nb, w.mln, RM);
  { GemmP g = gp0(); g.A=w.mln; g.B=P.qkv; g.D=w.bigA;
    g.M=RM; g.N=768; g.K=CC; g.sAm=CC; g.sAk=1; g.sBk=768; g.sBn=1; g.sDm=768;
    launch_gemm(st, g, 1); }
  { GemmP g = gp0();                                              // scores[h][l][i][j]
    g.A=w.bigA; g.B=w.bigA+256; g.D=w.genC;
    g.M=RR; g.N=RR; g.K=DMM;
    g.sAm=(long)LL*768; g.sAk=1; g.sBk=1; g.sBn=(long)LL*768; g.sDm=RR;
    g.nb2=LL;
    g.bA1=DMM; g.bA2=768; g.bB1=DMM; g.bB2=768;
    g.bD1=(long)LL*RR*RR; g.bD2=(long)RR*RR;
    launch_gemm(st, g, HMM); }
  { SoftP s{}; s.s=w.genC; s.bias=nullptr;
    s.n1=HMM; s.n2=LL; s.n3=RR; s.cols=RR;
    s.scale=1.f/sqrtf((float)DMM); launch_soft(st, s); }
  { GemmP g = gp0();                                              // out[(i*L+l), h*32+d]
    g.A=w.genC; g.B=w.bigA+512; g.D=w.genA;
    g.M=RR; g.N=DMM; g.K=RR;
    g.sAm=RR; g.sAk=1; g.sBk=(long)LL*768; g.sBn=1; g.sDm=(long)LL*256;
    g.nb2=LL;
    g.bA1=(long)LL*RR*RR; g.bA2=(long)RR*RR;
    g.bB1=DMM; g.bB2=768;
    g.bD1=DMM; g.bD2=256;
    launch_gemm(st, g, HMM); }
  { GemmP g = gp0(); g.A=w.mln; g.B=P.gw; g.bias=P.gb; g.D=w.genB; g.act=2;
    g.M=RM; g.N=256; g.K=CC; g.sAm=CC; g.sAk=1; g.sBk=256; g.sBn=1; g.sDm=256;
    launch_gemm(st, g, 1); }
  k_mul<<<pwgrid((long)RM*256), 256, 0, st>>>(w.genA, w.genA, w.genB, (long)RM*256);
  { GemmP g = gp0(); g.A=w.genA; g.B=P.ow; g.bias=P.ob; g.D=Mres; g.accum=1;
    g.M=RM; g.N=CC; g.K=256; g.sAm=256; g.sAk=1; g.sBk=CC; g.sBn=1; g.sDm=CC;
    launch_gemm(st, g, 1); }
}

static void op_transition(hipStream_t st, const TransP& P, float* X, long rows,
                          float* lnbuf, float* hid) {
  // chunk rows so hidden fits in genC (<= 16384 x 512)
  const long CH = 16384;
  k_ln<<<(unsigned)rows, 128, 0, st>>>(X, P.ng, P.nb, lnbuf, (int)rows);
  for (long r0 = 0; r0 < rows; r0 += CH) {
    long rc = rows - r0 < CH ? rows - r0 : CH;
    { GemmP g = gp0(); g.A=lnbuf + r0*CC; g.B=P.w1; g.bias=P.b1; g.D=hid; g.act=1;
      g.M=(int)rc; g.N=4*CC; g.K=CC; g.sAm=CC; g.sAk=1; g.sBk=4*CC; g.sBn=1; g.sDm=4*CC;
      launch_gemm(st, g, 1); }
    { GemmP g = gp0(); g.A=hid; g.B=P.w2; g.bias=P.b2; g.D=X + r0*CC; g.accum=1;
      g.M=(int)rc; g.N=CC; g.K=4*CC; g.sAm=4*CC; g.sAk=1; g.sBk=CC; g.sBn=1; g.sDm=CC;
      launch_gemm(st, g, 1); }
  }
}

static void op_opm(hipStream_t st, const OpmP& P, float* Mres, float* Zres, const WS& w) {
  const int DO = 32;
  k_ln<<<RM, 128, 0, st>>>(Mres, P.ng, P.nb, w.mln, RM);
  { GemmP g = gp0(); g.A=w.mln; g.B=P.w; g.bias=P.b; g.D=w.smallA;   // ab [RM, 64]
    g.M=RM; g.N=2*DO; g.K=CC; g.sAm=CC; g.sAk=1; g.sBk=2*DO; g.sBn=1; g.sDm=2*DO;
    launch_gemm(st, g, 1); }
  { CopyP c{}; c.src=w.smallA + DO; c.dst=w.smallB;                  // bp[r][j*32+d]
    c.n0=RR; c.n1=LL; c.n2=DO;
    c.ss0=(long)LL*2*DO; c.ss1=2*DO; c.ss2=1;
    c.ds0=(long)LL*DO;   c.ds1=DO;   c.ds2=1;
    k_copy3<<<pwgrid((long)RR*LL*DO), 256, 0, st>>>(c); }
  for (int ic = 0; ic < 8; ++ic) {        // chunks of 32 i-rows
    { GemmP g = gp0();                    // O[il][c][j][d] = (1/R) sum_r a b
      g.A=w.smallA + (long)ic*32*2*DO; g.B=w.smallB; g.D=w.genC;
      g.M=DO; g.N=LL*DO; g.K=RR;
      g.sAm=1; g.sAk=(long)LL*2*DO;
      g.sBk=(long)LL*DO; g.sBn=1;
      g.sDm=(long)LL*DO;
      g.bA1=2*DO; g.bD1=(long)DO*LL*DO;
      g.scale=1.f/(float)RR;
      launch_gemm(st, g, 32); }
    { GemmP g = gp0();                    // z += O @ ow + ob  (K = c x d via K2)
      g.A=w.genC; g.B=P.ow; g.bias=P.ob; g.D=Zres + (long)ic*32*LL*CC; g.accum=1;
      g.M=LL; g.N=CC; g.K=DO; g.K2=DO;
      g.sAm=DO; g.sAk=1; g.sAk2=(long)LL*DO;
      g.sBk=CC; g.sBk2=(long)DO*CC; g.sBn=1;
      g.sDm=CC;
      g.bA1=(long)DO*LL*DO; g.bD1=(long)LL*CC;
      launch_gemm(st, g, 32); }
  }
}

static void op_tri_mul(hipStream_t st, const TriMulP& P, float* Zres, const WS& w, int outDir) {
  const int DMUL = 128;
  k_ln<<<ZR, 128, 0, st>>>(Zres, P.ng, P.nb, w.zln, ZR);
  { GemmP g = gp0(); g.A=w.zln; g.B=P.w1; g.bias=P.b1; g.D=w.bigA;   // p1
    g.M=ZR; g.N=2*DMUL; g.K=CC; g.sAm=CC; g.sAk=1; g.sBk=2*DMUL; g.sBn=1; g.sDm=2*DMUL;
    launch_gemm(st, g, 1); }
  { GemmP g = gp0(); g.A=w.zln; g.B=P.w2; g.bias=P.b2; g.D=w.bigB; g.act=2; // sig(p2)
    g.M=ZR; g.N=2*DMUL; g.K=CC; g.sAm=CC; g.sAk=1; g.sBk=2*DMUL; g.sBn=1; g.sDm=2*DMUL;
    launch_gemm(st, g, 1); }
  k_mul<<<pwgrid((long)ZR*2*DMUL), 256, 0, st>>>(w.bigA, w.bigB, w.bigA, (long)ZR*2*DMUL);
  { GemmP g = gp0(); g.A=w.zln; g.B=P.gw; g.bias=P.gb; g.D=w.genB; g.act=2; // gate
    g.M=ZR; g.N=CC; g.K=CC; g.sAm=CC; g.sAk=1; g.sBk=CC; g.sBn=1; g.sDm=CC;
    launch_gemm(st, g, 1); }
  { GemmP g = gp0();                         // prod[i,j,d], batched over d
    g.A=w.bigA; g.B=w.bigA + DMUL; g.D=w.genA;
    g.M=LL; g.N=LL; g.K=LL;
    if (outDir) { g.sAm=(long)LL*2*DMUL; g.sAk=2*DMUL;
                  g.sBk=2*DMUL;          g.sBn=(long)LL*2*DMUL; }
    else        { g.sAm=2*DMUL;          g.sAk=(long)LL*2*DMUL;
                  g.sBk=(long)LL*2*DMUL; g.sBn=2*DMUL; }
    g.sDm=(long)LL*DMUL; g.sDn=DMUL;
    g.bA1=1; g.bB1=1; g.bD1=1;
    launch_gemm(st, g, DMUL); }
  k_ln<<<ZR, 128, 0, st>>>(w.genA, P.n2g, P.n2b, w.genC, ZR);
  { GemmP g = gp0(); g.A=w.genC; g.B=P.ow; g.bias=P.ob; g.D=w.genA;  // proj
    g.M=ZR; g.N=CC; g.K=DMUL; g.sAm=DMUL; g.sAk=1; g.sBk=CC; g.sBn=1; g.sDm=CC;
    launch_gemm(st, g, 1); }
  k_muladd<<<pwgrid((long)ZR*CC), 256, 0, st>>>(Zres, w.genB, w.genA, (long)ZR*CC);
}

static void op_tri_attn(hipStream_t st, const AttnP& P, float* Zres, const WS& w, int rowWise) {
  const int QKV = 3 * HPP * DPP;   // 384
  k_ln<<<ZR, 128, 0, st>>>(Zres, P.ng, P.nb, w.zln, ZR);
  { GemmP g = gp0(); g.A=w.zln; g.B=P.pair; g.D=w.smallA;            // bias [i,j,h]
    g.M=ZR; g.N=HPP; g.K=CC; g.sAm=CC; g.sAk=1; g.sBk=HPP; g.sBn=1; g.sDm=HPP;
    launch_gemm(st, g, 1); }
  { GemmP g = gp0(); g.A=w.zln; g.B=P.gw; g.bias=P.gb; g.D=w.genB; g.act=2; // gate
    g.M=ZR; g.N=CC; g.K=CC; g.sAm=CC; g.sAk=1; g.sBk=CC; g.sBn=1; g.sDm=CC;
    launch_gemm(st, g, 1); }
  for (int rc = 0; rc < 4; ++rc) {         // 4 chunks of 64 outer indices
    if (rowWise) {
      { GemmP g = gp0();                   // qkv chunk: contiguous 16384 rows
        g.A=w.zln + (long)rc*64*LL*CC; g.B=P.qkv; g.D=w.genC;
        g.M=64*LL; g.N=QKV; g.K=CC; g.sAm=CC; g.sAk=1; g.sBk=QKV; g.sBn=1; g.sDm=QKV;
        launch_gemm(st, g, 1); }
      { GemmP g = gp0();                   // scores[rl][h][i][j]
        g.A=w.genC; g.B=w.genC + HPP*DPP; g.D=w.bigB;
        g.M=LL; g.N=LL; g.K=DPP;
        g.sAm=QKV; g.sAk=1; g.sBk=1; g.sBn=QKV; g.sDm=LL;
        g.nb2=HPP;
        g.bA1=(long)LL*QKV; g.bA2=DPP;
        g.bB1=(long)LL*QKV; g.bB2=DPP;
        g.bD1=(long)HPP*LL*LL; g.bD2=(long)LL*LL;
        launch_gemm(st, g, 64); }
    } else {
      { GemmP g = gp0();                   // qkv chunk, layout [i][lc][384]
        g.A=w.zln + (long)rc*64*CC; g.B=P.qkv; g.D=w.genC;
        g.M=LL; g.N=QKV; g.K=CC;
        g.sAm=(long)LL*CC; g.sAk=1; g.sBk=QKV; g.sBn=1; g.sDm=(long)64*QKV;
        g.bA1=CC; g.bD1=QKV;
        launch_gemm(st, g, 64); }
      { GemmP g = gp0();                   // scores[lc][h][i][j]
        g.A=w.genC; g.B=w.genC + HPP*DPP; g.D=w.bigB;
        g.M=LL; g.N=LL; g.K=DPP;
        g.sAm=(long)64*QKV; g.sAk=1; g.sBk=1; g.sBn=(long)64*QKV; g.sDm=LL;
        g.nb2=HPP;
        g.bA1=QKV; g.bA2=DPP;
        g.bB1=QKV; g.bB2=DPP;
        g.bD1=(long)HPP*LL*LL; g.bD2=(long)LL*LL;
        launch_gemm(st, g, 64); }
    }
    { SoftP s{}; s.s=w.bigB; s.bias=w.smallA;         // bias depends on (i,j,h) only
      s.n1=64; s.n2=HPP; s.n3=LL; s.cols=LL;
      s.bs1=0; s.bs2=1; s.bs3=(long)LL*HPP; s.bsc=HPP;
      s.scale=1.f/sqrtf((float)DPP); launch_soft(st, s); }
    if (rowWise) {
      GemmP g = gp0();                     // out[(r*L+i), h*32+d]
      g.A=w.bigB; g.B=w.genC + 2*HPP*DPP; g.D=w.genA + (long)rc*64*LL*CC;
      g.M=LL; g.N=DPP; g.K=LL;
      g.sAm=LL; g.sAk=1; g.sBk=QKV; g.sBn=1; g.sDm=CC;
      g.nb2=HPP;
      g.bA1=(long)HPP*LL*LL; g.bA2=(long)LL*LL;
      g.bB1=(long)LL*QKV;    g.bB2=DPP;
      g.bD1=(long)LL*CC;     g.bD2=DPP;
      launch_gemm(st, g, 64);
    } else {
      GemmP g = gp0();                     // out[(i*L+l), h*32+d]
      g.A=w.bigB; g.B=w.genC + 2*HPP*DPP; g.D=w.genA + (long)rc*64*CC;
      g.M=LL; g.N=DPP; g.K=LL;
      g.sAm=LL; g.sAk=1; g.sBk=(long)64*QKV; g.sBn=1; g.sDm=(long)LL*CC;
      g.nb2=HPP;
      g.bA1=(long)HPP*LL*LL; g.bA2=(long)LL*LL;
      g.bB1=QKV;             g.bB2=DPP;
      g.bD1=CC;              g.bD2=DPP;
      launch_gemm(st, g, 64);
    }
  }
  k_mul<<<pwgrid((long)ZR*CC), 256, 0, st>>>(w.genA, w.genA, w.genB, (long)ZR*CC);
  { GemmP g = gp0(); g.A=w.genA; g.B=P.ow; g.bias=P.ob; g.D=Zres; g.accum=1;
    g.M=ZR; g.N=CC; g.K=HPP*DPP; g.sAm=HPP*DPP; g.sAk=1; g.sBk=CC; g.sBn=1; g.sDm=CC;
    launch_gemm(st, g, 1); }
}

// ---------------------------------------------------------------------------
extern "C" void kernel_launch(void* const* d_in, const int* in_sizes, int n_in,
                              void* d_out, int out_size, void* d_ws, size_t ws_size,
                              hipStream_t stream) {
  (void)n_in; (void)out_size; (void)ws_size;

  float* Mres = (float*)d_out;                 // [16384,128]
  float* Zres = Mres + (long)RM * CC;          // [65536,128]
  hipMemcpyAsync(Mres, d_in[0], (size_t)in_sizes[0] * sizeof(float),
                 hipMemcpyDeviceToDevice, stream);
  hipMemcpyAsync(Zres, d_in[1], (size_t)in_sizes[1] * sizeof(float),
                 hipMemcpyDeviceToDevice, stream);

  // Workspace layout (floats)
  WS w{};
  {
    float* base = (float*)d_ws; size_t o = 0;
    auto alloc = [&](size_t n) { float* p = base + o; o += n; return p; };
    w.zln    = alloc((size_t)ZR * CC);          //  8.4 M
    w.mln    = alloc((size_t)RM * CC);          //  2.1 M
    w.bigA   = alloc((size_t)ZR * 256);         // 16.8 M
    w.bigB   = alloc((size_t)ZR * 256);         // 16.8 M
    w.genA   = alloc((size_t)ZR * CC);          //  8.4 M
    w.genB   = alloc((size_t)ZR * CC);          //  8.4 M
    w.genC   = alloc((size_t)ZR * CC);          //  8.4 M
    w.smallA = alloc((size_t)RM * 64);          //  1.05 M
    w.smallB = alloc((size_t)RR * LL * 32);     //  0.52 M
  }

  // Parameter table (sorted-key flattening of each block dict)
  BlockP blocks[2];
  int ix = 2;
  auto nxt = [&]() { return (const float*)d_in[ix++]; };
  for (int b = 0; b < 2; ++b) {
    BlockP& B = blocks[b];
    B.col = { nxt(), nxt(), nxt(), nxt(), nxt(), nxt(), nxt() };
    B.mt  = { nxt(), nxt(), nxt(), nxt(), nxt(), nxt() };
    B.opm = { nxt(), nxt(), nxt(), nxt(), nxt(), nxt() };
    B.pt  = { nxt(), nxt(), nxt(), nxt(), nxt(), nxt() };
    B.row = { nxt(), nxt(), nxt(), nxt(), nxt(), nxt(), nxt(), nxt() };
    B.tca = { nxt(), nxt(), nxt(), nxt(), nxt(), nxt(), nxt(), nxt() };
    B.tmi = { nxt(), nxt(), nxt(), nxt(), nxt(), nxt(),
              nxt(), nxt(), nxt(), nxt(), nxt(), nxt() };
    B.tmo = { nxt(), nxt(), nxt(), nxt(), nxt(), nxt(),
              nxt(), nxt(), nxt(), nxt(), nxt(), nxt() };
    B.tra = { nxt(), nxt(), nxt(), nxt(), nxt(), nxt(), nxt(), nxt() };
  }

  for (int b = 0; b < 2; ++b) {
    const BlockP& P = blocks[b];
    op_msa_row(stream, P.row, Mres, Zres, w);
    op_msa_col(stream, P.col, Mres, w);
    op_transition(stream, P.mt, Mres, RM, w.mln, w.genC);
    op_opm(stream, P.opm, Mres, Zres, w);
    op_tri_mul(stream, P.tmo, Zres, w, 1);
    op_tri_mul(stream, P.tmi, Zres, w, 0);
    op_tri_attn(stream, P.tra, Zres, w, 1);
    op_tri_attn(stream, P.tca, Zres, w, 0);
    op_transition(stream, P.pt, Zres, ZR, w.zln, w.genC);
  }
}
